// LDGCNN_61538291417582
// MI455X (gfx1250) — compile-verified, matrix operand order checked by reading, not run on hardware
//
#include <hip/hip_runtime.h>

// ---------------- constants ----------------
#define B_   8
#define N_   2048
#define KNN  20
#define FTS  352          // f16 feature-transpose row stride (max padded C)
#define CTOT 323          // total concat channels in F
#define BNSC 0.9999950000374997f   // 1/sqrt(1+1e-5)

typedef __attribute__((ext_vector_type(16))) _Float16 v16h;
typedef __attribute__((ext_vector_type(8)))  _Float16 v8h;
typedef __attribute__((ext_vector_type(8)))  float    v8f;

union frag16 { v16h v; v8h h[2]; };

// CDNA5 16-bit A/B fragment layout per lane:
//   lanes 0-15 : e0..7 -> K 0..7,   e8..15 -> K 16..23
//   lanes16-31 : e0..7 -> K 8..15,  e8..15 -> K 24..31
// -> each lane's fragment = two contiguous 8xf16 (16B) runs: 128-bit loads.
__device__ __forceinline__ v16h load_frag(const _Float16* __restrict__ base,
                                          int f, int khalf) {
  frag16 u;
  u.h[0] = *(const v8h*)(base + f * 32 + khalf * 8);
  u.h[1] = *(const v8h*)(base + f * 32 + 16 + khalf * 8);
  return u.v;
}

// ---------------- small data-movement kernels ----------------
__global__ void copy_x_kernel(const float* __restrict__ x, float* __restrict__ F) {
  int i = blockIdx.x * blockDim.x + threadIdx.x;           // B*3*N
  if (i < B_ * 3 * N_) {
    int b = i / (3 * N_); int r = i % (3 * N_);
    F[(size_t)b * CTOT * N_ + r] = x[i];
  }
}

template<int C, int CP>
__global__ void transpose_f16_kernel(const float* __restrict__ F, _Float16* __restrict__ Ft) {
  int i = blockIdx.x * blockDim.x + threadIdx.x;           // B*N*CP
  if (i >= B_ * N_ * CP) return;
  int c = i % CP; int n = (i / CP) % N_; int b = i / (CP * N_);
  float v = (c < C) ? F[((size_t)b * CTOT + c) * N_ + n] : 0.0f;
  Ft[((size_t)b * N_ + n) * FTS + c] = (_Float16)v;
}

template<int O, int C, int CP>
__global__ void split_w_kernel(const float* __restrict__ w,
                               _Float16* __restrict__ Wa, _Float16* __restrict__ Wc) {
  int i = blockIdx.x * blockDim.x + threadIdx.x;           // O*CP
  if (i >= O * CP) return;
  int c = i % CP; int o = i / CP;
  if (c < C) {
    float a = w[o * 2 * C + c];
    float bb = w[o * 2 * C + C + c];
    Wa[i] = (_Float16)a;
    Wc[i] = (_Float16)(bb - a);
  } else {
    Wa[i] = (_Float16)0.0f;
    Wc[i] = (_Float16)0.0f;
  }
}

template<int O, int C, int CP>
__global__ void pack_w_kernel(const float* __restrict__ w, _Float16* __restrict__ W) {
  int i = blockIdx.x * blockDim.x + threadIdx.x;           // O*CP
  if (i >= O * CP) return;
  int c = i % CP; int o = i / CP;
  W[i] = (c < C) ? (_Float16)w[o * C + c] : (_Float16)0.0f;
}

// ---------------- kNN: per-point distance scan + 20x block argmin ----------------
template<int C>
__global__ __launch_bounds__(256) void knn_kernel(const float* __restrict__ F,
                                                  int* __restrict__ idx) {
  __shared__ float scen[224];
  __shared__ float dist[N_];
  __shared__ float rv[256];
  __shared__ int   ri[256];
  int bn = blockIdx.x; int b = bn / N_; int n = bn % N_;
  int tid = threadIdx.x;
  const float* Fb = F + (size_t)b * CTOT * N_;
  for (int c = tid; c < C; c += 256) scen[c] = Fb[(size_t)c * N_ + n];
  __syncthreads();
  for (int m = tid; m < N_; m += 256) {
    float d = 0.0f;
    for (int c = 0; c < C; ++c) {
      float t = Fb[(size_t)c * N_ + m] - scen[c];
      d = fmaf(t, t, d);
    }
    dist[m] = (m == n) ? 3.4e38f : d;   // exclude self
  }
  __syncthreads();
  for (int k = 0; k < KNN; ++k) {
    float bv = 3.4e38f; int bi = 0;
    for (int m = tid; m < N_; m += 256) {
      float v = dist[m];
      if (v < bv) { bv = v; bi = m; }
    }
    rv[tid] = bv; ri[tid] = bi;
    __syncthreads();
    for (int s = 128; s > 0; s >>= 1) {
      if (tid < s && rv[tid + s] < rv[tid]) { rv[tid] = rv[tid + s]; ri[tid] = ri[tid + s]; }
      __syncthreads();
    }
    if (tid == 0) {
      idx[((size_t)b * N_ + n) * KNN + k] = ri[0];
      dist[ri[0]] = 3.4e38f;
    }
    __syncthreads();
  }
}

// ---------------- edge conv: WMMA gather-GEMM fused with BN+LReLU+max_k ----------------
// 4 waves/block; wave w owns point tile n0 = (blockIdx.x*4 + w)*16.
// out[b,o,n] = max_k lrelu(bn( Wa . Ft[:,idx(n,k)] + Wc . Ft[:,n] ))
template<int CP, int O>
__global__ __launch_bounds__(128) void edge_conv_wmma(
    const _Float16* __restrict__ Wa, const _Float16* __restrict__ Wc,
    const _Float16* __restrict__ Ft, const int* __restrict__ idx,
    const float* __restrict__ g, const float* __restrict__ bb,
    float* __restrict__ F, int rowoff) {
  constexpr int NF = CP / 32;
  int lane = threadIdx.x & 31;
  int wave = threadIdx.x >> 5;
  int jn = lane & 15, khalf = lane >> 4;
  int n0 = (blockIdx.x * 4 + wave) * 16;
  int o0 = blockIdx.y * 16;
  int b  = blockIdx.z;

  // A fragments: weight rows (o0+jn), via two B128 loads per fragment
  v16h a_frag[NF], c_frag[NF];
  {
    const _Float16* wa = Wa + (size_t)(o0 + jn) * CP;
    const _Float16* wc = Wc + (size_t)(o0 + jn) * CP;
#pragma unroll
    for (int f = 0; f < NF; ++f) {
      a_frag[f] = load_frag(wa, f, khalf);
      c_frag[f] = load_frag(wc, f, khalf);
    }
  }

  const _Float16* ftb = Ft + (size_t)b * N_ * FTS;

  // center term (once per point tile): ct = Wc . Ft[:, n0+jn]
  v8f ct = {};
  {
    const _Float16* col = ftb + (size_t)(n0 + jn) * FTS;
#pragma unroll
    for (int f = 0; f < NF; ++f) {
      v16h bf = load_frag(col, f, khalf);
      ct = __builtin_amdgcn_wmma_f32_16x16x32_f16(false, c_frag[f], false, bf,
                                                  (short)0, ct, false, false);
    }
  }

  float scale[8], bias[8], mx[8];
#pragma unroll
  for (int r = 0; r < 8; ++r) {
    int o = o0 + r + khalf * 8;
    scale[r] = g[o] * BNSC;
    bias[r]  = bb[o];
    mx[r]    = -3.4e38f;
  }

  const int* idxb = idx + ((size_t)b * N_ + n0 + jn) * KNN;
  for (int k = 0; k < KNN; ++k) {
    int p = idxb[k];
    const _Float16* col = ftb + (size_t)p * FTS;
    v8f acc = ct;                                  // seed with center term
#pragma unroll
    for (int f = 0; f < NF; ++f) {
      v16h bf = load_frag(col, f, khalf);
      acc = __builtin_amdgcn_wmma_f32_16x16x32_f16(false, a_frag[f], false, bf,
                                                   (short)0, acc, false, false);
    }
#pragma unroll
    for (int r = 0; r < 8; ++r) {
      float v = fmaf(acc[r], scale[r], bias[r]);
      v = (v > 0.0f) ? v : 0.2f * v;
      mx[r] = fmaxf(mx[r], v);
    }
  }

  // write into concat buffer F rows [rowoff, rowoff+O)
#pragma unroll
  for (int r = 0; r < 8; ++r) {
    int o = o0 + r + khalf * 8;
    F[((size_t)b * CTOT + rowoff + o) * N_ + n0 + jn] = mx[r];
  }
}

// ---------------- conv5: WMMA GEMM + BN + LReLU + max over 16-point tile ----------------
template<int CP>
__global__ __launch_bounds__(128) void gemm5_wmma(
    const _Float16* __restrict__ W, const _Float16* __restrict__ Ft,
    const float* __restrict__ g, const float* __restrict__ bb,
    float* __restrict__ partial) {
  constexpr int NF = CP / 32;
  int lane = threadIdx.x & 31;
  int wave = threadIdx.x >> 5;
  int jn = lane & 15, khalf = lane >> 4;
  int tile = blockIdx.x * 4 + wave;
  int n0 = tile * 16;
  int o0 = blockIdx.y * 16;
  int b  = blockIdx.z;

  v16h a_frag[NF];
  {
    const _Float16* wr = W + (size_t)(o0 + jn) * CP;
#pragma unroll
    for (int f = 0; f < NF; ++f) a_frag[f] = load_frag(wr, f, khalf);
  }

  const _Float16* col = Ft + ((size_t)b * N_ + n0 + jn) * FTS;
  v8f acc = {};
#pragma unroll
  for (int f = 0; f < NF; ++f) {
    v16h bf = load_frag(col, f, khalf);
    acc = __builtin_amdgcn_wmma_f32_16x16x32_f16(false, a_frag[f], false, bf,
                                                 (short)0, acc, false, false);
  }

  float v[8];
#pragma unroll
  for (int r = 0; r < 8; ++r) {
    int o = o0 + r + khalf * 8;
    float t = fmaf(acc[r], g[o] * BNSC, bb[o]);
    v[r] = (t > 0.0f) ? t : 0.2f * t;
  }
  // max over the 16 columns (lanes within each 16-lane group)
#pragma unroll
  for (int off = 1; off < 16; off <<= 1)
#pragma unroll
    for (int r = 0; r < 8; ++r) v[r] = fmaxf(v[r], __shfl_xor(v[r], off, 32));

  if (jn == 0) {
#pragma unroll
    for (int r = 0; r < 8; ++r) {
      int o = o0 + r + khalf * 8;
      partial[((size_t)b * 1024 + o) * (N_ / 16) + tile] = v[r];
    }
  }
}

__global__ void reduce_max5_kernel(const float* __restrict__ partial, float* __restrict__ x5) {
  int i = blockIdx.x * blockDim.x + threadIdx.x;           // B*1024
  if (i >= B_ * 1024) return;
  const float* p = partial + (size_t)i * (N_ / 16);
  float m = p[0];
  for (int t = 1; t < N_ / 16; ++t) m = fmaxf(m, p[t]);
  x5[i] = m;
}

// ---------------- MLP head: one block per batch ----------------
__global__ __launch_bounds__(256) void head_kernel(
    const float* __restrict__ x5,
    const float* __restrict__ m1w, const float* __restrict__ m1b,
    const float* __restrict__ g6, const float* __restrict__ b6,
    const float* __restrict__ m2w, const float* __restrict__ m2b,
    const float* __restrict__ g7, const float* __restrict__ b7,
    const float* __restrict__ m3w, const float* __restrict__ m3b,
    float* __restrict__ out) {
  __shared__ float s5[1024];
  __shared__ float s6[256];
  __shared__ float s7[64];
  __shared__ float so[5];
  int b = blockIdx.x, tid = threadIdx.x;
  for (int i = tid; i < 1024; i += 256) s5[i] = x5[(size_t)b * 1024 + i];
  __syncthreads();
  {
    float acc = m1b[tid];
    for (int c = 0; c < 1024; ++c) acc = fmaf(m1w[(size_t)tid * 1024 + c], s5[c], acc);
    acc = fmaf(acc, g6[tid] * BNSC, b6[tid]);
    s6[tid] = (acc > 0.0f) ? acc : 0.2f * acc;
  }
  __syncthreads();
  if (tid < 64) {
    float acc = m2b[tid];
    for (int c = 0; c < 256; ++c) acc = fmaf(m2w[(size_t)tid * 256 + c], s6[c], acc);
    acc = fmaf(acc, g7[tid] * BNSC, b7[tid]);
    s7[tid] = (acc > 0.0f) ? acc : 0.2f * acc;
  }
  __syncthreads();
  if (tid < 5) {
    float acc = m3b[tid];
    for (int c = 0; c < 64; ++c) acc = fmaf(m3w[(size_t)tid * 64 + c], s7[c], acc);
    so[tid] = acc;
  }
  __syncthreads();
  if (tid == 0) {
    float nrm = 0.0f;
    for (int j = 0; j < 5; ++j) nrm += fabsf(so[j]);
    nrm = fmaxf(nrm, 1e-12f);
    for (int j = 0; j < 5; ++j) out[b * 5 + j] = so[j] / nrm;
  }
}

// ---------------- workspace layout ----------------
static constexpr size_t ALGN(size_t x) { return (x + 255) & ~(size_t)255; }
static constexpr size_t OFF_F   = 0;
static constexpr size_t SZ_F    = (size_t)B_ * CTOT * N_ * 4;
static constexpr size_t OFF_FT  = ALGN(OFF_F + SZ_F);
static constexpr size_t SZ_FT   = (size_t)B_ * N_ * FTS * 2;
static constexpr size_t OFF_IDX = ALGN(OFF_FT + SZ_FT);
static constexpr size_t SZ_IDX  = (size_t)B_ * N_ * KNN * 4;
static constexpr size_t OFF_WA  = ALGN(OFF_IDX + SZ_IDX);
static constexpr size_t SZ_W    = (size_t)1024 * FTS * 2;
static constexpr size_t OFF_WC  = ALGN(OFF_WA + SZ_W);
static constexpr size_t OFF_P5  = ALGN(OFF_WC + SZ_W);
static constexpr size_t SZ_P5   = (size_t)B_ * 1024 * (N_ / 16) * 4;
static constexpr size_t OFF_X5  = ALGN(OFF_P5 + SZ_P5);

// ---------------- per-layer driver ----------------
template<int C, int CP, int O>
static void run_layer(const float* w, const float* g, const float* bb,
                      float* F, _Float16* Ft, int* idx, _Float16* Wa, _Float16* Wc,
                      int rowoff, hipStream_t stream) {
  int tn = B_ * N_ * CP;
  transpose_f16_kernel<C, CP><<<(tn + 255) / 256, 256, 0, stream>>>(F, Ft);
  knn_kernel<C><<<B_ * N_, 256, 0, stream>>>(F, idx);
  int wn = O * CP;
  split_w_kernel<O, C, CP><<<(wn + 255) / 256, 256, 0, stream>>>(w, Wa, Wc);
  dim3 grid(N_ / 64, O / 16, B_);
  edge_conv_wmma<CP, O><<<grid, 128, 0, stream>>>(Wa, Wc, Ft, idx, g, bb, F, rowoff);
}

extern "C" void kernel_launch(void* const* d_in, const int* in_sizes, int n_in,
                              void* d_out, int out_size, void* d_ws, size_t ws_size,
                              hipStream_t stream) {
  (void)in_sizes; (void)n_in; (void)out_size; (void)ws_size;
  const float* x   = (const float*)d_in[0];
  const float* w1  = (const float*)d_in[1];
  const float* w2  = (const float*)d_in[2];
  const float* w3  = (const float*)d_in[3];
  const float* w4  = (const float*)d_in[4];
  const float* w5  = (const float*)d_in[5];
  const float* g1  = (const float*)d_in[6];  const float* b1 = (const float*)d_in[7];
  const float* g2  = (const float*)d_in[8];  const float* b2 = (const float*)d_in[9];
  const float* g3  = (const float*)d_in[10]; const float* b3 = (const float*)d_in[11];
  const float* g4  = (const float*)d_in[12]; const float* b4 = (const float*)d_in[13];
  const float* g5  = (const float*)d_in[14]; const float* b5 = (const float*)d_in[15];
  const float* g6  = (const float*)d_in[16]; const float* b6 = (const float*)d_in[17];
  const float* g7  = (const float*)d_in[18]; const float* b7 = (const float*)d_in[19];
  const float* m1w = (const float*)d_in[20]; const float* m1b = (const float*)d_in[21];
  const float* m2w = (const float*)d_in[22]; const float* m2b = (const float*)d_in[23];
  const float* m3w = (const float*)d_in[24]; const float* m3b = (const float*)d_in[25];

  char* ws = (char*)d_ws;
  float*    F   = (float*)(ws + OFF_F);
  _Float16* Ft  = (_Float16*)(ws + OFF_FT);
  int*      idx = (int*)(ws + OFF_IDX);
  _Float16* Wa  = (_Float16*)(ws + OFF_WA);
  _Float16* Wc  = (_Float16*)(ws + OFF_WC);
  float*    p5  = (float*)(ws + OFF_P5);
  float*    x5  = (float*)(ws + OFF_X5);

  int cn = B_ * 3 * N_;
  copy_x_kernel<<<(cn + 255) / 256, 256, 0, stream>>>(x, F);

  run_layer<3,   32,  64 >(w1, g1, b1, F, Ft, idx, Wa, Wc, 3,   stream);
  run_layer<67,  96,  64 >(w2, g2, b2, F, Ft, idx, Wa, Wc, 67,  stream);
  run_layer<131, 160, 64 >(w3, g3, b3, F, Ft, idx, Wa, Wc, 131, stream);
  run_layer<195, 224, 128>(w4, g4, b4, F, Ft, idx, Wa, Wc, 195, stream);

  // conv5 + global max + head
  int tn = B_ * N_ * 352;
  transpose_f16_kernel<323, 352><<<(tn + 255) / 256, 256, 0, stream>>>(F, Ft);
  int wn = 1024 * 352;
  pack_w_kernel<1024, 323, 352><<<(wn + 255) / 256, 256, 0, stream>>>(w5, Wa);
  dim3 g5grid(N_ / 64, 1024 / 16, B_);
  gemm5_wmma<352><<<g5grid, 128, 0, stream>>>(Wa, Ft, g5, b5, p5);
  int rn = B_ * 1024;
  reduce_max5_kernel<<<(rn + 255) / 256, 256, 0, stream>>>(p5, x5);
  head_kernel<<<B_, 256, 0, stream>>>(x5, m1w, m1b, g6, b6, m2w, m2b, g7, b7,
                                      m3w, m3b, (float*)d_out);
}
